// MultiHeadAttention_24223615549757
// MI455X (gfx1250) — matrix operand order repaired and checked by reading, not compiled, not run on hardware
//
#include <hip/hip_runtime.h>
#include <hip/hip_bf16.h>

#define NSEQ  4096
#define EDIM  2048
#define HEADS 16
#define DHEAD 128

typedef __attribute__((ext_vector_type(16))) __bf16 v16bf;
typedef __attribute__((ext_vector_type(8)))  __bf16 v8bf;
typedef __attribute__((ext_vector_type(8)))  float  v8f;

static __device__ __forceinline__ v16bf cat16(v8bf lo, v8bf hi) {
  v16bf r;
#pragma unroll
  for (int i = 0; i < 8; ++i) { r[i] = lo[i]; r[i + 8] = hi[i]; }
  return r;
}

static __device__ __forceinline__ v8f wmma_bf16(v16bf a, v16bf b, v8f c) {
  // D = A(16x32 bf16) * B(32x16 bf16) + C(16x16 f32)
  return __builtin_amdgcn_wmma_f32_16x16x32_bf16(false, a, false, b,
                                                 (short)0, c, false, false);
}

// Async global->LDS copy of 16B per lane (VDST = LDS byte address in VGPR,
// VADDR = 64-bit global address). Tracked on ASYNCcnt.
static __device__ __forceinline__ void async_ld16(const void* gptr,
                                                  unsigned ldsoff) {
  asm volatile("global_load_async_to_lds_b128 %0, %1, off"
               :: "v"(ldsoff), "v"(gptr) : "memory");
}
static __device__ __forceinline__ void wait_async0() {
  asm volatile("s_wait_asynccnt 0x0" ::: "memory");
}
// Flat shared-aperture pointers keep the LDS byte offset in bits [31:0].
static __device__ __forceinline__ unsigned lds_off(const void* p) {
  return (unsigned)(unsigned long long)p;
}

// ---------------------------------------------------------------------------
// fp32 -> bf16 conversion (grid-stride)
// ---------------------------------------------------------------------------
__global__ void k_f32_to_bf16(const float* __restrict__ in,
                              __bf16* __restrict__ out, int n) {
  int i = blockIdx.x * blockDim.x + threadIdx.x;
  int stride = gridDim.x * blockDim.x;
  for (; i < n; i += stride) out[i] = (__bf16)in[i];
}

// ---------------------------------------------------------------------------
// Tiled bf16 WMMA GEMM: C(f32, MxN) = A(bf16, MxK, row-major) * B(bf16, KxN)
// Block tile 128x128, K-step 32, 256 threads = 8 waves, 8 WMMA tiles/wave.
// A tile staged via GLOBAL_LOAD_ASYNC_TO_LDS_B128; B transposed via VGPRs.
// ---------------------------------------------------------------------------
__global__ __launch_bounds__(256) void k_gemm_bf16(
    const __bf16* __restrict__ A, const __bf16* __restrict__ B,
    float* __restrict__ C, int M, int N, int K) {
  __shared__ __bf16 Als[128][40];   // 128x32 tile, +8 pad keeps 16B alignment
  __shared__ __bf16 Bls[128][40];   // B tile stored transposed: [n][k]

  const int tid  = threadIdx.x;
  const int wave = tid >> 5;
  const int lane = tid & 31;
  const int bm = blockIdx.y * 128;
  const int bn = blockIdx.x * 128;
  const int m0 = (wave >> 1) * 32;   // 32-row band per wave pair
  const int n0 = (wave & 1) * 64;    // 64-col half
  const int arow  = lane & 15;
  const int koff  = (lane >> 4) * 8;   // unused here; A-frag also uses kbase
  const int kbase = (lane >> 4) * 16;  // B-frag K base (16 contiguous)

  v8f acc[2][4];
#pragma unroll
  for (int i = 0; i < 2; ++i)
#pragma unroll
    for (int j = 0; j < 4; ++j) acc[i][j] = (v8f){};

  const int lar = tid >> 1, lac = (tid & 1) * 16;   // A staging: 2 thr/row
  const int lbk = tid >> 3, lbc = (tid & 7) * 16;   // B staging: 8 thr/row
  const unsigned alds = lds_off(&Als[lar][lac]);

  for (int kb = 0; kb < K; kb += 32) {
    { // stage A 128x32 asynchronously (no VGPR round-trip)
      const __bf16* src = A + (size_t)(bm + lar) * K + kb + lac;
      async_ld16(src, alds);
      async_ld16(src + 8, alds + 16);
    }
    { // stage B 32x128, transposed into [n][k]
      const __bf16* src = B + (size_t)(kb + lbk) * N + bn + lbc;
      v8bf x0 = *(const v8bf*)src;
      v8bf x1 = *(const v8bf*)(src + 8);
#pragma unroll
      for (int e = 0; e < 8; ++e) Bls[lbc + e][lbk] = x0[e];
#pragma unroll
      for (int e = 0; e < 8; ++e) Bls[lbc + 8 + e][lbk] = x1[e];
    }
    if (kb + 32 < K) {
      __builtin_prefetch(A + (size_t)(bm + lar) * K + kb + 32 + lac, 0, 1);
      __builtin_prefetch(B + (size_t)(kb + 32 + lbk) * N + bn + lbc, 0, 1);
    }
    wait_async0();     // own async LDS writes landed before barrier signal
    __syncthreads();

    v16bf af[2], bfr[4];
#pragma unroll
    for (int mt = 0; mt < 2; ++mt) {
      const __bf16* p = &Als[m0 + mt * 16 + arow][0];
      af[mt] = cat16(*(const v8bf*)(p + kbase), *(const v8bf*)(p + kbase + 8));
    }
#pragma unroll
    for (int nt = 0; nt < 4; ++nt) {
      const __bf16* p = &Bls[n0 + nt * 16 + arow][0];
      bfr[nt] = cat16(*(const v8bf*)(p + kbase), *(const v8bf*)(p + kbase + 8));
    }
#pragma unroll
    for (int mt = 0; mt < 2; ++mt)
#pragma unroll
      for (int nt = 0; nt < 4; ++nt)
        acc[mt][nt] = wmma_bf16(af[mt], bfr[nt], acc[mt][nt]);
    __syncthreads();
  }

  const int rhalf = (lane >> 4) * 8;
#pragma unroll
  for (int mt = 0; mt < 2; ++mt)
#pragma unroll
    for (int nt = 0; nt < 4; ++nt)
#pragma unroll
      for (int r = 0; r < 8; ++r) {
        int row = bm + m0 + mt * 16 + r + rhalf;
        int col = bn + n0 + nt * 16 + arow;
        C[(size_t)row * N + col] = acc[mt][nt][r];
      }
}

// ---------------------------------------------------------------------------
// Fused RoPE + per-head LayerNorm; splits proj into bf16 q/k/v [H][N][D]
// One 128-thread block per (n, h).
// ---------------------------------------------------------------------------
static __device__ __forceinline__ float bsum128(float v, float* buf, int tid) {
  buf[tid] = v;
  __syncthreads();
#pragma unroll
  for (int s = 64; s > 0; s >>= 1) {
    if (tid < s) buf[tid] += buf[tid + s];
    __syncthreads();
  }
  float r = buf[0];
  __syncthreads();
  return r;
}

__global__ __launch_bounds__(128) void k_rope_ln(
    const float* __restrict__ proj, const float* __restrict__ ln_scale,
    const float* __restrict__ ln_bias, __bf16* __restrict__ qb,
    __bf16* __restrict__ kb, __bf16* __restrict__ vb) {
  const int n = blockIdx.x, h = blockIdx.y, d = threadIdx.x;
  __shared__ float qs[128], ks[128], red[128];

  const float* row = proj + (size_t)n * (3 * EDIM);
  float qv = row[h * DHEAD + d];
  float kv = row[EDIM + h * DHEAD + d];
  float vv = row[2 * EDIM + h * DHEAD + d];
  qs[d] = qv;
  ks[d] = kv;
  __syncthreads();

  const int j = d & 63;
  const float LOG_BASE = 9.210340371976184f;  // ln(10000)
  float ang = (float)n * __expf(-(2.0f * (float)j / 128.0f) * LOG_BASE);
  float c = __cosf(ang), s = __sinf(ang);
  float qr, kr;
  if (d < 64) {
    qr = qs[d] * c - qs[d + 64] * s;
    kr = ks[d] * c - ks[d + 64] * s;
  } else {
    qr = qs[d - 64] * s + qs[d] * c;
    kr = ks[d - 64] * s + ks[d] * c;
  }
  __syncthreads();

  // LayerNorm over D=128 for q and k
  float qm = bsum128(qr, red, d) * (1.0f / 128.0f);
  float qv2 = bsum128(qr * qr, red, d) * (1.0f / 128.0f) - qm * qm;
  float qn = (qr - qm) * rsqrtf(qv2 + 1e-5f) * ln_scale[d] + ln_bias[d];

  float km = bsum128(kr, red, d) * (1.0f / 128.0f);
  float kv2 = bsum128(kr * kr, red, d) * (1.0f / 128.0f) - km * km;
  float kn = (kr - km) * rsqrtf(kv2 + 1e-5f) * ln_scale[d] + ln_bias[d];

  size_t idx = ((size_t)h * NSEQ + n) * DHEAD + d;
  qb[idx] = (__bf16)qn;
  kb[idx] = (__bf16)kn;
  vb[idx] = (__bf16)vv;
}

// ---------------------------------------------------------------------------
// Causal degree-4 polynomial attention, flash-style, bf16 WMMA.
// Block = (128-row query stripe, head); 8 waves x 16 query rows each.
// K tile staged via async global->LDS; V transposed through VGPRs.
// ---------------------------------------------------------------------------
__global__ __launch_bounds__(256) void k_poly_attn(
    const __bf16* __restrict__ qb, const __bf16* __restrict__ kbuf,
    const __bf16* __restrict__ vbuf, __bf16* __restrict__ attn) {
  __shared__ __bf16 Kls[128][128];  // K tile (row-major); reused as P staging
  __shared__ __bf16 Vls[128][128];  // V tile transposed: [d][key]

  const int ib = blockIdx.x, h = blockIdx.y;
  const int tid = threadIdx.x, wave = tid >> 5, lane = tid & 31;
  const __bf16* Qh = qb + (size_t)h * NSEQ * DHEAD;
  const __bf16* Kh = kbuf + (size_t)h * NSEQ * DHEAD;
  const __bf16* Vh = vbuf + (size_t)h * NSEQ * DHEAD;

  const int row0  = ib * 128 + wave * 16;
  const int arow  = lane & 15;
  const int koff  = (lane >> 4) * 8;
  const int kbase = (lane >> 4) * 8;
  const int rhalf = (lane >> 4) * 8;

  // Q A-fragments (16 rows x 128 D = 4 k-steps), kept in registers
  v16bf qf[4];
#pragma unroll
  for (int ks = 0; ks < 4; ++ks) {
    const __bf16* p = Qh + (size_t)(row0 + arow) * DHEAD + ks * 32 + koff;
    qf[ks] = cat16(*(const v8bf*)p, *(const v8bf*)(p + 16));
  }

  v8f oacc[8];
  v8f dacc = (v8f){};
#pragma unroll
  for (int i = 0; i < 8; ++i) oacc[i] = (v8f){};

  const int lr = tid >> 1, lc = (tid & 1) * 64;  // staging: 2 threads per row
  const unsigned klds = lds_off(&Kls[lr][lc]);

  for (int jb = 0; jb <= ib; ++jb) {
    __syncthreads();  // protect Kls/Vls from readers of previous iteration
    {                 // stage K rows [jb*128, +128) async; V transposed
      const __bf16* src = Kh + (size_t)(jb * 128 + lr) * DHEAD + lc;
#pragma unroll
      for (int e = 0; e < 8; ++e) async_ld16(src + e * 8, klds + e * 16);
      const __bf16* vs = Vh + (size_t)(jb * 128 + lr) * DHEAD + lc;
#pragma unroll
      for (int e8 = 0; e8 < 8; ++e8) {
        v8bf v = *(const v8bf*)(vs + e8 * 8);
#pragma unroll
        for (int e = 0; e < 8; ++e) Vls[lc + e8 * 8 + e][lr] = v[e];
      }
    }
    wait_async0();
    __syncthreads();

    // S strip = Q * K^T  (16 x 128), then p = s^4, causal mask, row sums
    v8f sreg[8];
#pragma unroll
    for (int nt = 0; nt < 8; ++nt) {
      v8f s = (v8f){};
#pragma unroll
      for (int ks = 0; ks < 4; ++ks) {
        const __bf16* p = &Kls[nt * 16 + arow][ks * 32 + kbase];
        v16bf bf = cat16(*(const v8bf*)p, *(const v8bf*)(p + 16));
        s = wmma_bf16(qf[ks], bf, s);
      }
      const int colg = jb * 128 + nt * 16 + arow;
#pragma unroll
      for (int r = 0; r < 8; ++r) {
        int rowg = row0 + r + rhalf;
        float x = s[r];
        x = x * x;
        x = x * x;                 // degree 4
        if (colg > rowg) x = 0.f;  // causal
        s[r] = x;
        float t = x;  // row-sum across the 16-lane N group (stays in half)
        t += __shfl_xor(t, 1);
        t += __shfl_xor(t, 2);
        t += __shfl_xor(t, 4);
        t += __shfl_xor(t, 8);
        dacc[r] += t;
      }
      sreg[nt] = s;
    }
    __syncthreads();  // everyone done reading Kls; reuse it for P staging

    // write P strip (own 16 rows) as bf16, C-layout -> row-major LDS
#pragma unroll
    for (int nt = 0; nt < 8; ++nt)
#pragma unroll
      for (int r = 0; r < 8; ++r)
        Kls[wave * 16 + r + rhalf][nt * 16 + arow] = (__bf16)sreg[nt][r];

    // re-read own strip as A-fragments (same-wave DS ordering)
    v16bf pf[4];
#pragma unroll
    for (int ks = 0; ks < 4; ++ks) {
      const __bf16* p = &Kls[wave * 16 + arow][ks * 32 + koff];
      pf[ks] = cat16(*(const v8bf*)p, *(const v8bf*)(p + 16));
    }

    // O += P * V   (N dim = D, B-fragments contiguous from transposed V)
#pragma unroll
    for (int nt = 0; nt < 8; ++nt) {
#pragma unroll
      for (int ks = 0; ks < 4; ++ks) {
        const __bf16* p = &Vls[nt * 16 + arow][ks * 32 + kbase];
        v16bf bf = cat16(*(const v8bf*)p, *(const v8bf*)(p + 16));
        oacc[nt] = wmma_bf16(pf[ks], bf, oacc[nt]);
      }
    }
  }

  // normalize by denominator and store bf16 into [N][E]
#pragma unroll
  for (int nt = 0; nt < 8; ++nt)
#pragma unroll
    for (int r = 0; r < 8; ++r) {
      float o = oacc[nt][r] / dacc[r];
      int rowg = row0 + r + rhalf;
      int col = h * DHEAD + nt * 16 + arow;
      attn[(size_t)rowg * EDIM + col] = (__bf16)o;
    }
}

// ---------------------------------------------------------------------------
// Host-side launcher
// ---------------------------------------------------------------------------
extern "C" void kernel_launch(void* const* d_in, const int* in_sizes, int n_in,
                              void* d_out, int out_size, void* d_ws,
                              size_t ws_size, hipStream_t stream) {
  const float* x        = (const float*)d_in[0];
  const float* w_qkv    = (const float*)d_in[1];
  const float* w_out    = (const float*)d_in[2];
  const float* ln_scale = (const float*)d_in[3];
  const float* ln_bias  = (const float*)d_in[4];
  float* out = (float*)d_out;

  char* ws = (char*)d_ws;
  size_t off = 0;
  auto alloc = [&](size_t bytes) -> void* {
    void* p = ws + off;
    off += (bytes + 255) & ~(size_t)255;
    return p;
  };
  __bf16* x_bf    = (__bf16*)alloc((size_t)NSEQ * EDIM * 2);
  __bf16* wqkv_bf = (__bf16*)alloc((size_t)EDIM * 3 * EDIM * 2);
  __bf16* wout_bf = (__bf16*)alloc((size_t)EDIM * EDIM * 2);
  float*  proj    = (float*)alloc((size_t)NSEQ * 3 * EDIM * 4);
  __bf16* qb      = (__bf16*)alloc((size_t)HEADS * NSEQ * DHEAD * 2);
  __bf16* kb      = (__bf16*)alloc((size_t)HEADS * NSEQ * DHEAD * 2);
  __bf16* vb      = (__bf16*)alloc((size_t)HEADS * NSEQ * DHEAD * 2);
  __bf16* attn    = (__bf16*)alloc((size_t)NSEQ * EDIM * 2);

  k_f32_to_bf16<<<2048, 256, 0, stream>>>(x, x_bf, NSEQ * EDIM);
  k_f32_to_bf16<<<2048, 256, 0, stream>>>(w_qkv, wqkv_bf, EDIM * 3 * EDIM);
  k_f32_to_bf16<<<2048, 256, 0, stream>>>(w_out, wout_bf, EDIM * EDIM);

  dim3 g1(3 * EDIM / 128, NSEQ / 128);
  k_gemm_bf16<<<g1, 256, 0, stream>>>(x_bf, wqkv_bf, proj, NSEQ, 3 * EDIM, EDIM);

  dim3 g2(NSEQ, HEADS);
  k_rope_ln<<<g2, 128, 0, stream>>>(proj, ln_scale, ln_bias, qb, kb, vb);

  dim3 g3(NSEQ / 128, HEADS);
  k_poly_attn<<<g3, 256, 0, stream>>>(qb, kb, vb, attn);

  dim3 g4(EDIM / 128, NSEQ / 128);
  k_gemm_bf16<<<g4, 256, 0, stream>>>(attn, wout_bf, out, NSEQ, EDIM, EDIM);
}
